// GNN_5480378269923
// MI455X (gfx1250) — matrix operand order repaired and checked by reading, not compiled, not run on hardware
//
#include <hip/hip_runtime.h>
#include <hip/hip_bf16.h>

typedef __attribute__((ext_vector_type(16))) _Float16 v16h;
typedef __attribute__((ext_vector_type(8)))  float    v8f;

union Frag16 { v16h v; uint4 q[2]; };

#define N_NODES 50000
#define N_EDGES 800000
#define C_OUT   96
#define BN_EPS  1e-5f

// ---------------- utility kernels ----------------

__global__ void zero_f32(float* __restrict__ p, size_t n) {
    size_t i = (size_t)blockIdx.x * blockDim.x + threadIdx.x;
    if (i < n) p[i] = 0.0f;
}

__global__ void f32_to_f16(const float* __restrict__ in, _Float16* __restrict__ out, size_t n) {
    size_t i = (size_t)blockIdx.x * blockDim.x + threadIdx.x;
    if (i < n) out[i] = (_Float16)in[i];
}

// Wt[n*C + k] = W[k*96 + n]  (transpose + f16 convert; weight is [C, 96])
__global__ void transpose_w(const float* __restrict__ W, _Float16* __restrict__ Wt, int C) {
    int i = blockIdx.x * blockDim.x + threadIdx.x;
    if (i >= C * C_OUT) return;
    int k = i / C_OUT, n = i % C_OUT;
    Wt[(size_t)n * C + k] = (_Float16)W[(size_t)k * C_OUT + n];
}

// ---------------- degree / norm ----------------

__global__ void deg_init(float* __restrict__ deg, int n) {
    int i = blockIdx.x * blockDim.x + threadIdx.x;
    if (i < n) deg[i] = 1.0f;   // self-loop
}

__global__ void deg_accum(const long long* __restrict__ dst, float* __restrict__ deg, int e) {
    int i = blockIdx.x * blockDim.x + threadIdx.x;
    if (i < e) atomicAdd(&deg[(int)dst[i]], 1.0f);
}

__global__ void deg_to_dinv(float* __restrict__ deg, int n) {
    int i = blockIdx.x * blockDim.x + threadIdx.x;
    if (i < n) deg[i] = rsqrtf(deg[i]);   // in place: deg -> dinv
}

// ---------------- WMMA GEMM:  H[N,96] = A16[N,C] @ (Wt[96,C])^T ----------------
// Wt staged in LDS via async global->LDS copies; one wave per 16-row tile;
// 6 col-tiles of 16; K fully unrolled (C is a template parameter).
template <int C>
__global__ void gemm_wmma(const _Float16* __restrict__ A, const _Float16* __restrict__ Wt,
                          float* __restrict__ H, int nTiles) {
    constexpr int SP = C + 8;          // padded LDS row stride (halfwords) -> bank-conflict-free
    constexpr int KB = C / 32;         // K blocks
    __shared__ __align__(16) _Float16 sW[C_OUT * SP];

    // ---- async copy Wt[96][C] -> sW[96][SP] in 16B chunks ----
    constexpr int CPR = C / 8;                 // 16B chunks per row
    constexpr int CHUNKS = C_OUT * CPR;
    for (int i = threadIdx.x; i < CHUNKS; i += blockDim.x) {
        int n = i / CPR, kc = i - n * CPR;
        unsigned long long ga = (unsigned long long)(uintptr_t)(Wt + (size_t)n * C + kc * 8);
        unsigned lo = (unsigned)(uintptr_t)(sW + n * SP + kc * 8);   // LDS byte address
        asm volatile("global_load_async_to_lds_b128 %0, %1, off"
                     :: "v"(lo), "v"(ga) : "memory");
    }
    asm volatile("s_wait_asynccnt 0x0" ::: "memory");
    __syncthreads();

    int wave = (blockIdx.x * blockDim.x + threadIdx.x) >> 5;
    if (wave >= nTiles) return;            // uniform per wave
    int lane = threadIdx.x & 31;
    int m    = lane & 15;                  // row/col within tile
    int hi   = lane >> 4;                  // lane half
    int row0 = wave * 16;

    // ---- hoist all A fragments (ISA 16-bit A layout) ----
    const _Float16* arow = A + (size_t)(row0 + m) * C;
    Frag16 a[KB];
#pragma unroll
    for (int kb = 0; kb < KB; ++kb) {
        const _Float16* ap = arow + kb * 32 + hi * 8;
        a[kb].q[0] = *(const uint4*)ap;
        a[kb].q[1] = *(const uint4*)(ap + 16);
    }

    v8f acc[6];
#pragma unroll
    for (int ct = 0; ct < 6; ++ct) acc[ct] = (v8f){0.f,0.f,0.f,0.f,0.f,0.f,0.f,0.f};

#pragma unroll
    for (int kb = 0; kb < KB; ++kb) {
        // batch all 6 B fragments from LDS, then 6 back-to-back WMMAs
        Frag16 b[6];
#pragma unroll
        for (int ct = 0; ct < 6; ++ct) {
            const _Float16* bp = sW + (ct * 16 + m) * SP + kb * 32 + hi * 8;
            b[ct].q[0] = ((const uint4*)bp)[0];
            b[ct].q[1] = ((const uint4*)bp)[2];
        }
#pragma unroll
        for (int ct = 0; ct < 6; ++ct)
            acc[ct] = __builtin_amdgcn_wmma_f32_16x16x32_f16(
                false, a[kb].v, false, b[ct].v, (short)0, acc[ct], false, false);
    }

    // C/D layout: VGPR j -> row (hi*8 + j), col = lane%16
#pragma unroll
    for (int ct = 0; ct < 6; ++ct) {
#pragma unroll
        for (int j = 0; j < 8; ++j) {
            int r = row0 + hi * 8 + j;
            H[(size_t)r * C_OUT + ct * 16 + m] = acc[ct][j];
        }
    }
}

// ---------------- edge scatter: agg[dst] += h[src] * dinv[src]*dinv[dst] ----------------
// 32 lanes per edge, 3 channels per lane (96 = 3*32), lane-coalesced atomics (L2-resident).
__global__ void scatter_edges(const long long* __restrict__ src, const long long* __restrict__ dst,
                              const float* __restrict__ dinv, const float* __restrict__ H,
                              float* __restrict__ agg, int e) {
    int g = blockIdx.x * blockDim.x + threadIdx.x;
    int ed = g >> 5;
    if (ed >= e) return;
    int lane = g & 31;
    int s = (int)src[ed], d = (int)dst[ed];
    float norm = dinv[s] * dinv[d];
    const float* hs = H   + (size_t)s * C_OUT;
    float*       ad = agg + (size_t)d * C_OUT;
#pragma unroll
    for (int c = lane; c < C_OUT; c += 32)
        atomicAdd(&ad[c], hs[c] * norm);
}

// ---------------- BN pass A: y = agg + h*dinv^2 + bias (in place over agg), accumulate sums ----
// block = 96 threads (one per channel), 64 rows per block.
__global__ void bn_stats(float* __restrict__ yagg, const float* __restrict__ H,
                         const float* __restrict__ dinv, const float* __restrict__ bias,
                         float* __restrict__ sums /* [2*96]: sum, sumsq */, int n) {
    int c  = threadIdx.x;
    int r0 = blockIdx.x * 64;
    int r1 = r0 + 64; if (r1 > n) r1 = n;
    float b = bias[c];
    float s = 0.f, s2 = 0.f;
    for (int i = r0; i < r1; ++i) {
        float di = dinv[i];
        size_t idx = (size_t)i * C_OUT + c;
        float y = yagg[idx] + H[idx] * di * di + b;
        yagg[idx] = y;
        s += y; s2 += y * y;
    }
    atomicAdd(&sums[c], s);
    atomicAdd(&sums[C_OUT + c], s2);
}

// ---------------- BN pass B: out = relu((y-m)*rsqrt(v+eps)*g + beta); optional mean-pool accum --
__global__ void bn_apply(const float* __restrict__ y, const float* __restrict__ sums,
                         const float* __restrict__ gamma, const float* __restrict__ beta,
                         float* __restrict__ out, float* __restrict__ pool, int n, int do_pool) {
    int c = threadIdx.x;
    float m   = sums[c] / (float)n;
    float var = sums[C_OUT + c] / (float)n - m * m;
    float sc  = rsqrtf(var + BN_EPS) * gamma[c];
    float bt  = beta[c];
    int r0 = blockIdx.x * 64;
    int r1 = r0 + 64; if (r1 > n) r1 = n;
    float p = 0.f;
    for (int i = r0; i < r1; ++i) {
        size_t idx = (size_t)i * C_OUT + c;
        float v = (y[idx] - m) * sc + bt;
        v = v > 0.f ? v : 0.f;
        out[idx] = v;
        p += v;
    }
    if (do_pool) atomicAdd(&pool[c], p);
}

// ---------------- MLP head (single block, 128 threads) ----------------
__global__ void mlp_head(const float* __restrict__ pool,
                         const float* __restrict__ fw1, const float* __restrict__ fb1,
                         const float* __restrict__ fw2, const float* __restrict__ fb2,
                         const float* __restrict__ fw3, const float* __restrict__ fb3,
                         const float* __restrict__ fw4, const float* __restrict__ fb4,
                         float* __restrict__ out, int n) {
    __shared__ float v0[96], v1[128], v2[64], v3[32];
    int t = threadIdx.x;
    if (t < 96) v0[t] = pool[t] / (float)n;
    __syncthreads();
    if (t < 128) {
        float s = fb1[t];
        for (int c = 0; c < 96; ++c) s += v0[c] * fw1[(size_t)c * 128 + t];
        v1[t] = fmaxf(s, 0.f);
    }
    __syncthreads();
    if (t < 64) {
        float s = fb2[t];
        for (int c = 0; c < 128; ++c) s += v1[c] * fw2[(size_t)c * 64 + t];
        v2[t] = fmaxf(s, 0.f);
    }
    __syncthreads();
    if (t < 32) {
        float s = fb3[t];
        for (int c = 0; c < 64; ++c) s += v2[c] * fw3[(size_t)c * 32 + t];
        v3[t] = fmaxf(s, 0.f);
    }
    __syncthreads();
    if (t == 0) {
        float s = fb4[0];
        for (int c = 0; c < 32; ++c) s += v3[c] * fw4[c];
        out[0] = s;
    }
}

// ---------------- host orchestration ----------------

extern "C" void kernel_launch(void* const* d_in, const int* in_sizes, int n_in,
                              void* d_out, int out_size, void* d_ws, size_t ws_size,
                              hipStream_t stream) {
    (void)in_sizes; (void)n_in; (void)out_size; (void)ws_size;

    const float* x   = (const float*)d_in[0];
    const long long* ei = (const long long*)d_in[1];
    const long long* esrc = ei;
    const long long* edst = ei + N_EDGES;
    const float* W[3]  = { (const float*)d_in[2], (const float*)d_in[4], (const float*)d_in[6] };
    const float* B[3]  = { (const float*)d_in[3], (const float*)d_in[5], (const float*)d_in[7] };
    const float* G[3]  = { (const float*)d_in[8], (const float*)d_in[10], (const float*)d_in[12] };
    const float* BT[3] = { (const float*)d_in[9], (const float*)d_in[11], (const float*)d_in[13] };
    const float* fw1 = (const float*)d_in[14]; const float* fb1 = (const float*)d_in[15];
    const float* fw2 = (const float*)d_in[16]; const float* fb2 = (const float*)d_in[17];
    const float* fw3 = (const float*)d_in[18]; const float* fb3 = (const float*)d_in[19];
    const float* fw4 = (const float*)d_in[20]; const float* fb4 = (const float*)d_in[21];
    float* out = (float*)d_out;

    // workspace carve-up (256B aligned)
    char* ws = (char*)d_ws;
    size_t off = 0;
    auto carve = [&](size_t bytes) {
        char* p = ws + off;
        off += (bytes + 255) & ~(size_t)255;
        return p;
    };
    _Float16* xh   = (_Float16*)carve((size_t)N_NODES * 128 * sizeof(_Float16));
    _Float16* wt   = (_Float16*)carve((size_t)C_OUT * 128 * sizeof(_Float16));
    float*    h    = (float*)carve((size_t)N_NODES * C_OUT * sizeof(float));
    float*    agg  = (float*)carve((size_t)N_NODES * C_OUT * sizeof(float));
    float*    dinv = (float*)carve((size_t)N_NODES * sizeof(float));
    float*    sums = (float*)carve((size_t)3 * C_OUT * sizeof(float)); // sum, sumsq, pool

    const int nTiles   = N_NODES / 16;                 // 3125
    const int gemmGrid = (nTiles + 3) / 4;             // 4 waves/block (128 thr)
    const int bnGrid   = (N_NODES + 63) / 64;

    // degrees -> dinv (in place)
    deg_init<<<(N_NODES + 255) / 256, 256, 0, stream>>>(dinv, N_NODES);
    deg_accum<<<(N_EDGES + 255) / 256, 256, 0, stream>>>(edst, dinv, N_EDGES);
    deg_to_dinv<<<(N_NODES + 255) / 256, 256, 0, stream>>>(dinv, N_NODES);

    const float* fin = x;
    for (int l = 0; l < 3; ++l) {
        int C = (l == 0) ? 128 : 96;
        size_t nf = (size_t)N_NODES * C;

        // convert features + weights to f16 (weights transposed)
        f32_to_f16<<<(int)((nf + 255) / 256), 256, 0, stream>>>(fin, xh, nf);
        transpose_w<<<(C * C_OUT + 255) / 256, 256, 0, stream>>>(W[l], wt, C);

        // zero agg and stats (sum, sumsq, pool)
        zero_f32<<<(int)(((size_t)N_NODES * C_OUT + 255) / 256), 256, 0, stream>>>(agg, (size_t)N_NODES * C_OUT);
        zero_f32<<<1, 3 * C_OUT, 0, stream>>>(sums, 3 * C_OUT);

        // dense GEMM (WMMA, LDS-staged weights) : h = xh @ W
        if (C == 128)
            gemm_wmma<128><<<gemmGrid, 128, 0, stream>>>(xh, wt, h, nTiles);
        else
            gemm_wmma<96><<<gemmGrid, 128, 0, stream>>>(xh, wt, h, nTiles);

        // sym-normalized scatter-add over edges
        scatter_edges<<<(int)(((size_t)N_EDGES * 32 + 255) / 256), 256, 0, stream>>>(
            esrc, edst, dinv, h, agg, N_EDGES);

        // y = agg + h*dinv^2 + bias (in place over agg) + channel sums
        bn_stats<<<bnGrid, C_OUT, 0, stream>>>(agg, h, dinv, B[l], sums, N_NODES);

        // BN + ReLU -> write into h (becomes next layer input); layer 3 also mean-pool sums
        bn_apply<<<bnGrid, C_OUT, 0, stream>>>(agg, sums, G[l], BT[l], h,
                                               sums + 2 * C_OUT, N_NODES, (l == 2) ? 1 : 0);
        fin = h;
    }

    // MLP head on pooled vector
    mlp_head<<<1, 128, 0, stream>>>(sums + 2 * C_OUT, fw1, fb1, fw2, fb2, fw3, fb3, fw4, fb4,
                                    out, N_NODES);
}
